// DualGraphEncoder_54571854463225
// MI455X (gfx1250) — compile-verified, matrix operand order checked by reading, not run on hardware
//
#include <hip/hip_runtime.h>
#include <math.h>

#define DEVINL __device__ __forceinline__

typedef _Float16     v16h __attribute__((ext_vector_type(16)));
typedef __fp16       fph2 __attribute__((ext_vector_type(2)));
typedef float        v8f  __attribute__((ext_vector_type(8)));
typedef unsigned int u4v  __attribute__((ext_vector_type(4)));

union HFrag { v16h v; unsigned int u[8]; u4v q[2]; };

constexpr int Nn   = 4096;
constexpr int Ee   = 131072;
constexpr int Dd   = 256;
constexpr int Ll   = 2;
constexpr int MSDd = 512;
constexpr int GHh  = 128;
// SCALE = 1/sqrt(64) = 0.125

// k-index mapping for 16-bit WMMA A/B fragments (16x32):
// dword i of the fragment holds elements K = kmap(i,half), kmap(i,half)+1
DEVINL int kmap(int i, int half) { return ((i & 4) << 2) + (half << 3) + ((i & 3) << 1); }

DEVINL v8f wmma_f16(v16h a, v16h b, v8f c) {
  return __builtin_amdgcn_wmma_f32_16x16x32_f16(false, a, false, b, (short)0, c, false, false);
}

// pack two f32 -> one dword of two f16 (v_cvt_pk_rtz_f16_f32)
DEVINL unsigned int pkh(float a, float b) {
  fph2 r = __builtin_amdgcn_cvt_pkrtz(a, b);
  union { fph2 h; unsigned int u; } cv; cv.h = r;
  return cv.u;
}

// Load a 16x32 f16 WMMA fragment straight from f16 global memory:
// each half of the fragment is 16 contiguous bytes -> 2 x global_load_b128.
DEVINL void load_frag_h(const _Float16* __restrict__ base, int half, HFrag& f) {
  f.q[0] = *(const u4v*)(base + 8 * half);
  f.q[1] = *(const u4v*)(base + 16 + 8 * half);
}

// ---------------------------------------------------------------------------
// Graph scatter-mean
// ---------------------------------------------------------------------------
__global__ void k_count(const int* __restrict__ rows, float* __restrict__ cnt) {
  int i = blockIdx.x * 256 + threadIdx.x;
  atomicAdd(&cnt[rows[i]], 1.0f);
}

__global__ void k_scatter(const int* __restrict__ rows, const int* __restrict__ cols,
                          const float* __restrict__ src, int lds_src,
                          float* __restrict__ acc) {
  int e = blockIdx.x;
  int d = threadIdx.x;
  int r = rows[e], c = cols[e];
  atomicAdd(&acc[r * Dd + d], src[c * lds_src + d]);
}

__global__ void k_divcnt(float* __restrict__ nei, const float* __restrict__ cnt) {
  int idx = blockIdx.x * 256 + threadIdx.x;
  int n = idx >> 8;
  nei[idx] = nei[idx] / (cnt[n] + 1e-12f);
}

// ---------------------------------------------------------------------------
// f32 [M][ldin] (width W = 8<<wsh) -> compact f16 [M][W]. 8 elems/thread.
// ---------------------------------------------------------------------------
__global__ __launch_bounds__(256) void k_cvt16s(const float* __restrict__ in, int ldin,
                                                _Float16* __restrict__ out, int wsh) {
  int idx8 = blockIdx.x * 256 + threadIdx.x;
  int row = idx8 >> wsh;
  int c8 = (idx8 & ((1 << wsh) - 1)) * 8;
  int W = (1 << wsh) * 8;
  const float4* p = (const float4*)&in[row * ldin + c8];
  float4 a = p[0], b = p[1];
  u4v u;
  u.x = pkh(a.x, a.y); u.y = pkh(a.z, a.w);
  u.z = pkh(b.x, b.y); u.w = pkh(b.z, b.w);
  *(u4v*)&out[row * W + c8] = u;
}

// ---------------------------------------------------------------------------
// Weight f32 [K][N] -> transposed f16 [N][K] (K/4 = 1<<ksh), 4x4 micro-tiles.
// ---------------------------------------------------------------------------
__global__ __launch_bounds__(256) void k_cvtw_t(const float* __restrict__ Win,
                                                _Float16* __restrict__ Wt,
                                                int K, int N, int ksh) {
  int t = blockIdx.x * 256 + threadIdx.x;
  int kg = t & ((1 << ksh) - 1), ng = t >> ksh;
  int k0 = kg * 4, n0 = ng * 4;
  float rr[4][4];
#pragma unroll
  for (int j = 0; j < 4; ++j) {
    float4 f = *(const float4*)&Win[(k0 + j) * N + n0];
    rr[j][0] = f.x; rr[j][1] = f.y; rr[j][2] = f.z; rr[j][3] = f.w;
  }
#pragma unroll
  for (int jn = 0; jn < 4; ++jn) {
    uint2 u;
    u.x = pkh(rr[0][jn], rr[1][jn]);
    u.y = pkh(rr[2][jn], rr[3][jn]);
    *(uint2*)&Wt[(n0 + jn) * K + k0] = u;
  }
}

// ---------------------------------------------------------------------------
// LDS-free WMMA GEMM:  C = act( A1@W1 [+ A2@W2] + bias1 [+ bias2] [+ resid] )
// A: compact f16 [M][K]; W: pre-transposed f16 [N][K].
// Block: 256 thr (8 waves), tile 64 x 128; fragments load direct from global.
// ---------------------------------------------------------------------------
__global__ __launch_bounds__(256) void k_gemm_h(
    const _Float16* __restrict__ A1, const _Float16* __restrict__ W1,
    const _Float16* __restrict__ A2, const _Float16* __restrict__ W2,
    const float* __restrict__ bias1, const float* __restrict__ bias2,
    const float* __restrict__ resid, float* __restrict__ Cout, int ldc,
    int K, int Nout, int act) {
  const int tid  = threadIdx.x;
  const int w    = tid >> 5, lane = tid & 31;
  const int half = lane >> 4, nl = lane & 15;
  const int rstrip = w >> 1, cgrp = w & 1;
  const int mblock = blockIdx.x * 64, nblock = blockIdx.y * 128;

  v8f acc[4] = {};
  const int nph = (A2 != nullptr) ? 2 : 1;

  for (int ph = 0; ph < nph; ++ph) {
    const _Float16* Ap = ph ? A2 : A1;
    const _Float16* Wp = ph ? W2 : W1;
    const _Float16* abase = Ap + (mblock + rstrip * 16 + nl) * K;
    const _Float16* bbase = Wp + (nblock + cgrp * 64 + nl) * K;

    for (int kw = 0; kw < K; kw += 32) {
      if (kw + 32 < K) __builtin_prefetch(abase + kw + 32, 0, 3);
      HFrag fa;
      load_frag_h(abase + kw, half, fa);
#pragma unroll
      for (int c = 0; c < 4; ++c) {
        HFrag fb;
        load_frag_h(bbase + c * 16 * K + kw, half, fb);
        acc[c] = wmma_f16(fa.v, fb.v, acc[c]);
      }
    }
  }

#pragma unroll
  for (int c = 0; c < 4; ++c) {
    int col = nblock + cgrp * 64 + c * 16 + nl;
    float bb = 0.0f;
    if (bias1) bb += bias1[col];
    if (bias2) bb += bias2[col];
#pragma unroll
    for (int r = 0; r < 8; ++r) {
      int row = mblock + rstrip * 16 + r + 8 * half;
      float vv = acc[c][r] + bb;
      if (resid) vv += resid[row * ldc + col];
      if (act == 1) vv = fmaxf(vv, 0.0f);
      Cout[row * ldc + col] = vv;
    }
  }
}

// ---------------------------------------------------------------------------
// Row LayerNorm (+ optional exact GELU). One block per row.
// ---------------------------------------------------------------------------
__global__ __launch_bounds__(256) void k_ln(
    const float* __restrict__ in, int ldin,
    const float* __restrict__ g, const float* __restrict__ b,
    float* __restrict__ out, int ldout, int W, int act) {
  const int row = blockIdx.x, t = threadIdx.x;
  __shared__ float red[256];
  float s = 0.0f, ss = 0.0f;
  for (int j = t; j < W; j += 256) {
    float x = in[row * ldin + j];
    s += x; ss += x * x;
  }
  red[t] = s; __syncthreads();
  for (int o = 128; o > 0; o >>= 1) { if (t < o) red[t] += red[t + o]; __syncthreads(); }
  float mean = red[0] / (float)W;
  __syncthreads();
  red[t] = ss; __syncthreads();
  for (int o = 128; o > 0; o >>= 1) { if (t < o) red[t] += red[t + o]; __syncthreads(); }
  float var = red[0] / (float)W - mean * mean;
  float inv = rsqrtf(var + 1e-5f);
  for (int j = t; j < W; j += 256) {
    float x = (in[row * ldin + j] - mean) * inv * g[j] + b[j];
    if (act == 2) x = 0.5f * x * (1.0f + erff(x * 0.70710678118f));
    out[row * ldout + j] = x;
  }
}

// ---------------------------------------------------------------------------
// Fusion helpers
// ---------------------------------------------------------------------------
__global__ void k_concat(const float* __restrict__ hs, const float* __restrict__ ha,
                         float* __restrict__ cn) {
  int idx = blockIdx.x * 256 + threadIdx.x;
  int n = idx >> 10, j = idx & 1023;
  cn[idx] = (j < MSDd) ? hs[n * MSDd + j] : ha[n * MSDd + (j - MSDd)];
}

__global__ __launch_bounds__(128) void k_gate2(const float* __restrict__ gh,
                                               const float* __restrict__ W2,
                                               const float* __restrict__ b2,
                                               float* __restrict__ wg) {
  int n = blockIdx.x, t = threadIdx.x;
  __shared__ float red[128];
  red[t] = gh[n * GHh + t] * W2[t];
  __syncthreads();
  for (int o = 64; o > 0; o >>= 1) { if (t < o) red[t] += red[t + o]; __syncthreads(); }
  if (t == 0) wg[n] = 1.0f / (1.0f + __expf(-(red[0] + b2[0])));
}

__global__ void k_fuse(const float* __restrict__ wg, const float* __restrict__ hs,
                       const float* __restrict__ ha, float* __restrict__ fused) {
  int idx = blockIdx.x * 256 + threadIdx.x;
  int n = idx >> 9;
  float ww = wg[n];
  fused[idx] = ww * hs[idx] + (1.0f - ww) * ha[idx];
}

// ---------------------------------------------------------------------------
// Flash talking-heads attention, pass A: per gate-head row max / logsumexp
// q/k are pre-converted f16 [N][256]. Block: 16 q-rows, 8 waves over key tiles.
// ---------------------------------------------------------------------------
__global__ __launch_bounds__(256) void k_attn_stats(
    const _Float16* __restrict__ q, const _Float16* __restrict__ kmat,
    const float* __restrict__ Wpre, const float* __restrict__ bpre,
    float* __restrict__ Mst, float* __restrict__ Lst) {
  const int tid = threadIdx.x;
  const int w = tid >> 5, lane = tid & 31;
  const int half = lane >> 4, nl = lane & 15;
  const int rowbase = blockIdx.x * 16;

  float wpre_s[4][4];
#pragma unroll
  for (int h = 0; h < 4; ++h)
#pragma unroll
    for (int g = 0; g < 4; ++g) wpre_s[h][g] = Wpre[h * 4 + g] * 0.125f;
  float bp[4];
#pragma unroll
  for (int g = 0; g < 4; ++g) bp[g] = bpre[g];

  HFrag qa[4][2];
#pragma unroll
  for (int h = 0; h < 4; ++h)
#pragma unroll
    for (int ks = 0; ks < 2; ++ks)
      load_frag_h(q + (rowbase + nl) * Dd + h * 64 + ks * 32, half, qa[h][ks]);

  float mrun[4][8], lrun[4][8];
#pragma unroll
  for (int g = 0; g < 4; ++g)
#pragma unroll
    for (int r = 0; r < 8; ++r) { mrun[g][r] = -1e30f; lrun[g][r] = 0.0f; }

  for (int mt = w; mt < Nn / 16; mt += 8) {
    const int mbase = mt * 16;
    const _Float16* kb_base = kmat + (mbase + nl) * Dd;
    __builtin_prefetch(kb_base + 8 * 16 * Dd, 0, 3);
    v8f sc[4];
#pragma unroll
    for (int h = 0; h < 4; ++h) {
      v8f c = {};
#pragma unroll
      for (int ks = 0; ks < 2; ++ks) {
        HFrag kb;
        load_frag_h(kb_base + h * 64 + ks * 32, half, kb);
        c = wmma_f16(qa[h][ks].v, kb.v, c);
      }
      sc[h] = c;
    }
#pragma unroll
    for (int r = 0; r < 8; ++r) {
#pragma unroll
      for (int g = 0; g < 4; ++g) {
        float s = bp[g] + sc[0][r] * wpre_s[0][g] + sc[1][r] * wpre_s[1][g]
                        + sc[2][r] * wpre_s[2][g] + sc[3][r] * wpre_s[3][g];
        float mo = mrun[g][r];
        float mn = fmaxf(mo, s);
        lrun[g][r] = lrun[g][r] * __expf(mo - mn) + __expf(s - mn);
        mrun[g][r] = mn;
      }
    }
  }

  // merge across the 16 lanes holding each row
#pragma unroll
  for (int g = 0; g < 4; ++g)
#pragma unroll
    for (int r = 0; r < 8; ++r) {
      float m = mrun[g][r], l = lrun[g][r];
#pragma unroll
      for (int mask = 1; mask < 16; mask <<= 1) {
        float mo = __shfl_xor(m, mask, 32);
        float lo = __shfl_xor(l, mask, 32);
        float mn = fmaxf(m, mo);
        l = l * __expf(m - mn) + lo * __expf(mo - mn);
        m = mn;
      }
      mrun[g][r] = m; lrun[g][r] = l;
    }

  __shared__ float stM[8][4][16];
  __shared__ float stL[8][4][16];
  if (nl == 0) {
#pragma unroll
    for (int g = 0; g < 4; ++g)
#pragma unroll
      for (int r = 0; r < 8; ++r) {
        stM[w][g][r + 8 * half] = mrun[g][r];
        stL[w][g][r + 8 * half] = lrun[g][r];
      }
  }
  __syncthreads();
  if (tid < 64) {
    int g = tid >> 4, row = tid & 15;
    float m = stM[0][g][row], l = stL[0][g][row];
#pragma unroll
    for (int ww = 1; ww < 8; ++ww) {
      float mo = stM[ww][g][row], lo = stL[ww][g][row];
      float mn = fmaxf(m, mo);
      l = l * __expf(m - mn) + lo * __expf(mo - mn);
      m = mn;
    }
    Mst[g * Nn + rowbase + row] = m;
    Lst[g * Nn + rowbase + row] = l;
  }
}

__global__ void k_colsum(const float* __restrict__ v, float* __restrict__ cs) {
  int d = threadIdx.x;
  float s = 0.0f;
  for (int n = 0; n < Nn; ++n) s += v[n * Dd + d];
  cs[d] = s;
}

// ---------------------------------------------------------------------------
// Flash talking-heads attention, pass B: output accumulation.
// Block: 32 q-rows; wave (rh,g2) owns 16 rows x output head g2.
// ---------------------------------------------------------------------------
__global__ __launch_bounds__(256) void k_attn_out(
    const _Float16* __restrict__ q, const _Float16* __restrict__ kmat,
    const float* __restrict__ vmat,
    const float* __restrict__ Wpre, const float* __restrict__ bpre,
    const float* __restrict__ Wpost, const float* __restrict__ bpost,
    const float* __restrict__ Mst, const float* __restrict__ Lst,
    const float* __restrict__ colsum, float* __restrict__ o) {
  __shared__ __align__(16) _Float16 ldsV[256 * 36];   // V tile transposed: [d][m]
  __shared__ __align__(16) _Float16 stg[8][16 * 36];  // per-wave prob staging: [n][m]
  const int tid = threadIdx.x;
  const int w = tid >> 5, lane = tid & 31;
  const int half = lane >> 4, nl = lane & 15;
  const int g2 = w & 3, rh = w >> 2;
  const int rowbase = blockIdx.x * 32 + rh * 16;

  float wpre_s[4][4];
#pragma unroll
  for (int h = 0; h < 4; ++h)
#pragma unroll
    for (int g = 0; g < 4; ++g) wpre_s[h][g] = Wpre[h * 4 + g] * 0.125f;
  float bp[4];
#pragma unroll
  for (int g = 0; g < 4; ++g) bp[g] = bpre[g];

  float Mr[4][8], coef[4][8];
#pragma unroll
  for (int g = 0; g < 4; ++g)
#pragma unroll
    for (int r = 0; r < 8; ++r) {
      int row = rowbase + r + 8 * half;
      Mr[g][r]   = Mst[g * Nn + row];
      coef[g][r] = Wpost[g * 4 + g2] / Lst[g * Nn + row];
    }

  HFrag qa[4][2];
#pragma unroll
  for (int h = 0; h < 4; ++h)
#pragma unroll
    for (int ks = 0; ks < 2; ++ks)
      load_frag_h(q + (rowbase + nl) * Dd + h * 64 + ks * 32, half, qa[h][ks]);

  v8f oa[4] = {};

  for (int m0 = 0; m0 < Nn; m0 += 32) {
    __syncthreads();
    // ---- stage V tile 32x256 -> f16 transposed [d][m], 4x4 micro-tiles ----
#pragma unroll
    for (int j = 0; j < 2; ++j) {
      int idx = j * 256 + tid;            // 512 micro-tiles: 8 (m) x 64 (d)
      int dg = idx & 63, mg = idx >> 6;
      int m0l = mg * 4, d0 = dg * 4;
      float rr[4][4];
#pragma unroll
      for (int jm = 0; jm < 4; ++jm) {
        float4 t4 = *(const float4*)&vmat[(m0 + m0l + jm) * Dd + d0];
        rr[jm][0] = t4.x; rr[jm][1] = t4.y; rr[jm][2] = t4.z; rr[jm][3] = t4.w;
      }
#pragma unroll
      for (int jd = 0; jd < 4; ++jd) {
        uint2 u;
        u.x = pkh(rr[0][jd], rr[1][jd]);
        u.y = pkh(rr[2][jd], rr[3][jd]);
        *(uint2*)&ldsV[(d0 + jd) * 36 + m0l] = u;
      }
    }
    __syncthreads();

#pragma unroll
    for (int s = 0; s < 2; ++s) {
      const int mbase = m0 + s * 16;
      const _Float16* kb_base = kmat + (mbase + nl) * Dd;
      v8f sc[4];
#pragma unroll
      for (int h = 0; h < 4; ++h) {
        v8f c = {};
#pragma unroll
        for (int ks = 0; ks < 2; ++ks) {
          HFrag kb;
          load_frag_h(kb_base + h * 64 + ks * 32, half, kb);
          c = wmma_f16(qa[h][ks].v, kb.v, c);
        }
        sc[h] = c;
      }
#pragma unroll
      for (int r = 0; r < 8; ++r) {
        float pprob = 0.0f;
#pragma unroll
        for (int g = 0; g < 4; ++g) {
          float sv = bp[g] + sc[0][r] * wpre_s[0][g] + sc[1][r] * wpre_s[1][g]
                           + sc[2][r] * wpre_s[2][g] + sc[3][r] * wpre_s[3][g];
          pprob += coef[g][r] * __expf(sv - Mr[g][r]);
        }
        stg[w][(r + 8 * half) * 36 + s * 16 + nl] = (_Float16)pprob;
      }
    }
    __syncthreads();

    HFrag fa;
#pragma unroll
    for (int i = 0; i < 8; ++i)
      fa.u[i] = *(const unsigned int*)&stg[w][nl * 36 + kmap(i, half)];
#pragma unroll
    for (int c = 0; c < 4; ++c) {
      HFrag fb;
#pragma unroll
      for (int i = 0; i < 8; ++i)
        fb.u[i] = *(const unsigned int*)&ldsV[(g2 * 64 + c * 16 + nl) * 36 + kmap(i, half)];
      oa[c] = wmma_f16(fa.v, fb.v, oa[c]);
    }
  }

  float bp2 = bpost[g2];
#pragma unroll
  for (int c = 0; c < 4; ++c)
#pragma unroll
    for (int r = 0; r < 8; ++r) {
      int row = rowbase + r + 8 * half;
      int d = g2 * 64 + c * 16 + nl;
      o[row * Dd + d] = oa[c][r] + bp2 * colsum[d];
    }
}

// ---------------------------------------------------------------------------
__global__ __launch_bounds__(256) void k_l2norm(float* __restrict__ z) {
  int row = blockIdx.x, t = threadIdx.x;
  __shared__ float red[256];
  float x = z[row * Dd + t];
  red[t] = x * x;
  __syncthreads();
  for (int o = 128; o > 0; o >>= 1) { if (t < o) red[t] += red[t + o]; __syncthreads(); }
  float inv = 1.0f / fmaxf(sqrtf(red[0]), 1e-12f);
  z[row * Dd + t] = x * inv;
}

// ---------------------------------------------------------------------------
struct Ws16 {
  _Float16* ah;   // activations, up to N x 1024 halves
  _Float16* a2h;  // second activation, N x 256 halves
  _Float16* wt1;  // transposed weight, up to 131072 halves
  _Float16* wt2;  // second transposed weight, 65536 halves
};

// helper: convert + transpose + launch one GEMM (A f32 [M][ldin], W f32 [K][N])
static void gemm16(const float* A1, int lda1, int wsh1, const float* W1,
                   const float* A2, const float* W2,
                   const float* bias1, const float* bias2, const float* resid,
                   float* Cout, int ldc, int K, int Nout, int act,
                   const Ws16& t, hipStream_t stream) {
  int ksh = (K == 256) ? 6 : (K == 512 ? 7 : 8);
  k_cvt16s<<<16 << wsh1, 256, 0, stream>>>(A1, lda1, t.ah, wsh1);
  k_cvtw_t<<<(K / 4) * (Nout / 4) / 256, 256, 0, stream>>>(W1, t.wt1, K, Nout, ksh);
  if (A2) {
    k_cvt16s<<<16 << 5, 256, 0, stream>>>(A2, Dd, t.a2h, 5);
    k_cvtw_t<<<(K / 4) * (Nout / 4) / 256, 256, 0, stream>>>(W2, t.wt2, K, Nout, ksh);
  }
  k_gemm_h<<<dim3(Nn / 64, Nout / 128 ? Nout / 128 : 1), 256, 0, stream>>>(
      t.ah, t.wt1, A2 ? t.a2h : nullptr, A2 ? t.wt2 : nullptr,
      bias1, bias2, resid, Cout, ldc, K, Nout, act);
}

static void run_stream(const float* x, const int* edge,
                       const float* selfW, const float* selfb,
                       const float* neiW, const float* neib,
                       const float* lng, const float* lnb,
                       float* streambuf, float* nei, float* cnt, float* tmp,
                       const Ws16& t, hipStream_t stream) {
  const float* hprev = x;
  int ldh = Dd;
  for (int i = 0; i < Ll; ++i) {
    (void)hipMemsetAsync(nei, 0, (size_t)Nn * Dd * sizeof(float), stream);
    (void)hipMemsetAsync(cnt, 0, (size_t)Nn * sizeof(float), stream);
    k_count<<<Ee / 256, 256, 0, stream>>>(edge, cnt);
    k_scatter<<<Ee, 256, 0, stream>>>(edge, edge + Ee, hprev, ldh, nei);
    k_divcnt<<<(Nn * Dd) / 256, 256, 0, stream>>>(nei, cnt);
    gemm16(hprev, ldh, 5, selfW + (size_t)i * Dd * Dd,
           nei, neiW + (size_t)i * Dd * Dd,
           selfb + (size_t)i * Dd, neib + (size_t)i * Dd, nullptr,
           tmp, Dd, Dd, Dd, 0, t, stream);
    k_ln<<<Nn, 256, 0, stream>>>(tmp, Dd, lng + (size_t)i * Dd, lnb + (size_t)i * Dd,
                                 streambuf + (size_t)i * Dd, MSDd, Dd, 2);
    hprev = streambuf + (size_t)i * Dd;
    ldh = MSDd;
  }
}

extern "C" void kernel_launch(void* const* d_in, const int* in_sizes, int n_in,
                              void* d_out, int out_size, void* d_ws, size_t ws_size,
                              hipStream_t stream) {
  (void)in_sizes; (void)n_in; (void)out_size; (void)ws_size;
  const float* x       = (const float*)d_in[0];
  const int*   e_sp    = (const int*)d_in[1];
  const int*   e_at    = (const int*)d_in[2];
  const float* s_selfW = (const float*)d_in[3];
  const float* s_selfb = (const float*)d_in[4];
  const float* s_neiW  = (const float*)d_in[5];
  const float* s_neib  = (const float*)d_in[6];
  const float* s_lng   = (const float*)d_in[7];
  const float* s_lnb   = (const float*)d_in[8];
  const float* a_selfW = (const float*)d_in[9];
  const float* a_selfb = (const float*)d_in[10];
  const float* a_neiW  = (const float*)d_in[11];
  const float* a_neib  = (const float*)d_in[12];
  const float* a_lng   = (const float*)d_in[13];
  const float* a_lnb   = (const float*)d_in[14];
  const float* nc_g    = (const float*)d_in[15];
  const float* nc_b    = (const float*)d_in[16];
  const float* gate_W1 = (const float*)d_in[17];
  const float* gate_b1 = (const float*)d_in[18];
  const float* gate_W2 = (const float*)d_in[19];
  const float* gate_b2 = (const float*)d_in[20];
  const float* red_W   = (const float*)d_in[21];
  const float* red_b   = (const float*)d_in[22];
  const float* aln_g   = (const float*)d_in[23];
  const float* aln_b   = (const float*)d_in[24];
  const float* Wq      = (const float*)d_in[25];
  const float* Wk      = (const float*)d_in[26];
  const float* Wv      = (const float*)d_in[27];
  const float* Wo      = (const float*)d_in[28];
  const float* bq      = (const float*)d_in[29];
  const float* bk      = (const float*)d_in[30];
  const float* bv      = (const float*)d_in[31];
  const float* bo      = (const float*)d_in[32];
  const float* Wpre    = (const float*)d_in[33];
  const float* bpre    = (const float*)d_in[34];
  const float* Wpost   = (const float*)d_in[35];
  const float* bpost   = (const float*)d_in[36];

  float* ws = (float*)d_ws;
  size_t off = 0;
  auto alloc = [&](size_t n) { float* p = ws + off; off += n; return p; };
  float* nei   = alloc((size_t)Nn * Dd);
  float* cnt   = alloc(Nn);
  float* tmp   = alloc((size_t)Nn * Dd);
  float* hs    = alloc((size_t)Nn * MSDd);
  float* ha    = alloc((size_t)Nn * MSDd);
  float* cn    = alloc((size_t)Nn * 2 * MSDd);
  float* gh    = alloc((size_t)Nn * GHh);
  float* wg    = alloc(Nn);
  float* fused = alloc((size_t)Nn * MSDd);
  float* z     = alloc((size_t)Nn * Dd);
  float* xn    = alloc((size_t)Nn * Dd);
  float* qb    = alloc((size_t)Nn * Dd);
  float* kb    = alloc((size_t)Nn * Dd);
  float* vb    = alloc((size_t)Nn * Dd);
  float* ob    = alloc((size_t)Nn * Dd);
  float* Mst   = alloc((size_t)4 * Nn);
  float* Lst   = alloc((size_t)4 * Nn);
  float* cs    = alloc(Dd);
  Ws16 t;
  t.ah  = (_Float16*)alloc((size_t)Nn * 512);   // N x 1024 halves
  t.a2h = (_Float16*)alloc((size_t)Nn * 128);   // N x 256 halves
  t.wt1 = (_Float16*)alloc(65536);              // 131072 halves
  t.wt2 = (_Float16*)alloc(32768);              // 65536 halves
  _Float16* qh = (_Float16*)alloc((size_t)Nn * 128);
  _Float16* kh = (_Float16*)alloc((size_t)Nn * 128);

  // dual-stream multi-scale SAGE
  run_stream(x, e_sp, s_selfW, s_selfb, s_neiW, s_neib, s_lng, s_lnb,
             hs, nei, cnt, tmp, t, stream);
  run_stream(x, e_at, a_selfW, a_selfb, a_neiW, a_neib, a_lng, a_lnb,
             ha, nei, cnt, tmp, t, stream);

  // gated fusion
  k_concat<<<(Nn * 2 * MSDd) / 256, 256, 0, stream>>>(hs, ha, cn);
  k_ln<<<Nn, 256, 0, stream>>>(cn, 2 * MSDd, nc_g, nc_b, cn, 2 * MSDd, 2 * MSDd, 0);
  gemm16(cn, 2 * MSDd, 7, gate_W1, nullptr, nullptr,
         gate_b1, nullptr, nullptr, gh, GHh, 2 * MSDd, GHh, 1, t, stream);
  k_gate2<<<Nn, 128, 0, stream>>>(gh, gate_W2, gate_b2, wg);
  k_fuse<<<(Nn * MSDd) / 256, 256, 0, stream>>>(wg, hs, ha, fused);
  gemm16(fused, MSDd, 6, red_W, nullptr, nullptr,
         red_b, nullptr, nullptr, z, Dd, MSDd, Dd, 0, t, stream);

  // talking-heads attention
  k_ln<<<Nn, 256, 0, stream>>>(z, Dd, aln_g, aln_b, xn, Dd, Dd, 0);
  gemm16(xn, Dd, 5, Wq, nullptr, nullptr, bq, nullptr, nullptr, qb, Dd, Dd, Dd, 0, t, stream);
  gemm16(xn, Dd, 5, Wk, nullptr, nullptr, bk, nullptr, nullptr, kb, Dd, Dd, Dd, 0, t, stream);
  gemm16(xn, Dd, 5, Wv, nullptr, nullptr, bv, nullptr, nullptr, vb, Dd, Dd, Dd, 0, t, stream);
  k_cvt16s<<<16 << 5, 256, 0, stream>>>(qb, Dd, qh, 5);
  k_cvt16s<<<16 << 5, 256, 0, stream>>>(kb, Dd, kh, 5);
  k_attn_stats<<<Nn / 16, 256, 0, stream>>>(qh, kh, Wpre, bpre, Mst, Lst);
  k_colsum<<<1, 256, 0, stream>>>(vb, cs);
  k_attn_out<<<Nn / 32, 256, 0, stream>>>(qh, kh, vb, Wpre, bpre, Wpost, bpost,
                                          Mst, Lst, cs, ob);

  // residual output projection + L2 normalize
  float* zout = (float*)d_out;
  gemm16(ob, Dd, 5, Wo, nullptr, nullptr, bo, nullptr, z, zout, Dd, Dd, Dd, 0, t, stream);
  k_l2norm<<<Nn, 256, 0, stream>>>(zout);
}
